// ConvNextBlock_56238301774028
// MI455X (gfx1250) — matrix operand order repaired and checked by reading, not compiled, not run on hardware
//
#include <hip/hip_runtime.h>

// ---------------------------------------------------------------------------
// ConvNeXt-style sparse block on MI455X (gfx1250, wave32, WMMA bf16).
// conv1 (27-offset gather-GEMM) -> BN -> 1x1 (64->256) -> ReLU -> 1x1 (256->64) + x
// All GEMMs via v_wmma_f32_16x16x32_bf16 (fp32 accumulate).
// conv1: 2 M-tiles/wave (B-fragment reuse) + double-buffered A gathers.
// ---------------------------------------------------------------------------

#define NPTS   150000
#define DCH    64
#define KOFF   27
#define NTILES (NPTS / 16)        // 9375 (exact)
#define NPAIR  ((NTILES + 1) / 2) // 4688 tile-pairs (last pair half-valid)
#define WPB    8                  // waves per 256-thread block

typedef __bf16 bf16_t;
typedef __attribute__((ext_vector_type(16))) __bf16 v16bf;
typedef __attribute__((ext_vector_type(8)))  __bf16 v8bf;
typedef __attribute__((ext_vector_type(8)))  float  v8f;
typedef unsigned int u32;
typedef __attribute__((ext_vector_type(4)))  u32    v4u;

// Workspace layout (bytes, every segment 256B-aligned). Total ~57.9 MB.
#define WS_OUT1 0u           // f32  [N][64]   conv1 output (pre-BN)
#define WS_XB   38400000u    // bf16 [N][64]   x converted, row-major (gather source)
#define WS_W1T  57600000u    // bf16 [27][64][64] W1 column-major per offset
#define WS_W2T  57821184u    // bf16 [256][64] W2 column-major
#define WS_W3T  57853952u    // bf16 [64][256] W3 column-major
#define WS_STAT 57886720u    // f32  sum[64] | sumsq[64] | scale[64] | shift[64]

__device__ __forceinline__ v8f wmma_bf16(v16bf a, v16bf b, v8f c) {
  return __builtin_amdgcn_wmma_f32_16x16x32_bf16(false, a, false, b, (short)0, c,
                                                 false, false);
}

__device__ __forceinline__ v16bf concat8(v8bf lo, v8bf hi) {
  return __builtin_shufflevector(lo, hi, 0, 1, 2, 3, 4, 5, 6, 7,
                                 8, 9, 10, 11, 12, 13, 14, 15);
}

__device__ __forceinline__ v8f zero8() {
  v8f z;
#pragma unroll
  for (int j = 0; j < 8; ++j) z[j] = 0.f;
  return z;
}

// A-tile: 16x64 bf16 activations in WMMA A-layout = two K-steps of 32.
struct ATile { v16bf ks0, ks1; };

// Gather one 16-row A-tile for offset k (lane row = m, lane half h holds channels 16h..16h+15 of each K-step).
__device__ __forceinline__ void load_a_tile(const bf16_t* __restrict__ xb,
                                            const int* __restrict__ nidx,
                                            const int* __restrict__ nmask,
                                            int k, int m, int laneHi, ATile& t) {
  int gi = nidx[k * NPTS + m];
  u32 sel = nmask[k * NPTS + m] ? 0xFFFFFFFFu : 0u;
  v4u selv; selv[0] = sel; selv[1] = sel; selv[2] = sel; selv[3] = sel;
  const bf16_t* rowp = xb + (size_t)gi * DCH;
  int c0 = laneHi * 16;
  v4u a00 = *(const v4u*)(rowp + c0)      & selv;   // ks0: ch c0..c0+7
  v4u a01 = *(const v4u*)(rowp + c0 + 8)  & selv;   // ks0: ch c0+8..c0+15
  v4u a10 = *(const v4u*)(rowp + c0 + 32) & selv;   // ks1
  v4u a11 = *(const v4u*)(rowp + c0 + 40) & selv;
  t.ks0 = concat8(__builtin_bit_cast(v8bf, a00), __builtin_bit_cast(v8bf, a01));
  t.ks1 = concat8(__builtin_bit_cast(v8bf, a10), __builtin_bit_cast(v8bf, a11));
}

// One offset's GEMM burst: preload all 8 B fragments into distinct registers,
// then 16 WMMAs (2 M-tiles x 4 N-tiles x 2 K-steps). Distinct B registers let
// the scheduler overlap the load stream with the matrix pipe (partial waits).
__device__ __forceinline__ void conv1_step(const bf16_t* __restrict__ W1t, int k,
                                           int laneLo, int laneHi,
                                           const ATile& a0, const ATile& a1,
                                           v8f acc0[4], v8f acc1[4]) {
  const bf16_t* wk = W1t + (k << 12);
  v16bf b[8];
#pragma unroll
  for (int nt = 0; nt < 4; ++nt) {
    int col = nt * 16 + laneLo;
    b[2 * nt]     = *(const v16bf*)(wk + (col << 6) + laneHi * 16);
    b[2 * nt + 1] = *(const v16bf*)(wk + (col << 6) + 32 + laneHi * 16);
  }
#pragma unroll
  for (int nt = 0; nt < 4; ++nt) {
    acc0[nt] = wmma_bf16(a0.ks0, b[2 * nt],     acc0[nt]);
    acc0[nt] = wmma_bf16(a0.ks1, b[2 * nt + 1], acc0[nt]);
    acc1[nt] = wmma_bf16(a1.ks0, b[2 * nt],     acc1[nt]);
    acc1[nt] = wmma_bf16(a1.ks1, b[2 * nt + 1], acc1[nt]);
  }
}

// ---------------------------------------------------------------------------
// prep_x: x (f32) -> xb (bf16), row-major. One gathered row = 128B = 1 L2 line.
// ---------------------------------------------------------------------------
__global__ __launch_bounds__(256) void prep_x(const float* __restrict__ x,
                                              bf16_t* __restrict__ xb, int total) {
  int i = blockIdx.x * 256 + threadIdx.x;
  if (i < total) xb[i] = (bf16_t)x[i];
}

// ---------------------------------------------------------------------------
// prep_w: weights -> bf16 column-major (so each lane's WMMA B-fragment is a
// contiguous 32-byte load). Also zero the BN accumulators (graph-replay safe).
// ---------------------------------------------------------------------------
__global__ __launch_bounds__(256) void prep_w(const float* __restrict__ W1,
                                              const float* __restrict__ W2,
                                              const float* __restrict__ W3,
                                              bf16_t* __restrict__ W1t,
                                              bf16_t* __restrict__ W2t,
                                              bf16_t* __restrict__ W3t,
                                              float* __restrict__ stats) {
  int i = blockIdx.x * 256 + threadIdx.x;
  if (i < KOFF * DCH * DCH) {            // W1[k][r][c] -> W1t[k][c][r]
    int k = i >> 12, rem = i & 4095, c = rem >> 6, r = rem & 63;
    W1t[(k << 12) + (c << 6) + r] = (bf16_t)W1[(k << 12) + (r << 6) + c];
  }
  if (i < 256 * 64) {
    { int c = i >> 6, r = i & 63;  W2t[(c << 6) + r] = (bf16_t)W2[(r << 8) + c]; }
    { int c = i >> 8, r = i & 255; W3t[(c << 8) + r] = (bf16_t)W3[(r << 6) + c]; }
  }
  if (i < 2 * DCH) stats[i] = 0.f;
}

// ---------------------------------------------------------------------------
// conv1: one wave per PAIR of 16-row tiles (32 rows). Per offset: 8 B-b128-pair
// loads feed 16 WMMAs; A gathers for offset k+1 are issued before the WMMA
// burst for offset k (ping-pong double buffering). Mask applied branchlessly.
// Also accumulates BN sum/sumsq via LDS -> global atomics.
// ---------------------------------------------------------------------------
__global__ __launch_bounds__(256) void conv1_kernel(
    const bf16_t* __restrict__ xb, const int* __restrict__ nidx,
    const int* __restrict__ nmask, const bf16_t* __restrict__ W1t,
    float* __restrict__ out1, float* __restrict__ stats) {
  __shared__ float lsum[DCH];
  __shared__ float lssq[DCH];
  int tid = threadIdx.x;
  if (tid < DCH) { lsum[tid] = 0.f; lssq[tid] = 0.f; }
  __syncthreads();

  int wave = tid >> 5, lane = tid & 31;
  int laneLo = lane & 15, laneHi = lane >> 4;
  int pair = blockIdx.x * WPB + wave;

  if (pair < NPAIR) {                    // wave-uniform guard: EXEC stays all-1s
    int tile0 = pair * 2;
    bool t1valid = (tile0 + 1) < NTILES; // wave-uniform
    int row0 = tile0 << 4;
    int m0 = row0 + laneLo;                          // rows of tile 0
    int m1 = t1valid ? (row0 + 16 + laneLo) : m0;    // rows of tile 1 (safe dup)

    v8f acc0[4], acc1[4];
#pragma unroll
    for (int nt = 0; nt < 4; ++nt) { acc0[nt] = zero8(); acc1[nt] = zero8(); }

    ATile a0A, a1A, a0B, a1B;            // ping-pong A buffers (2 M-tiles each)
    load_a_tile(xb, nidx, nmask, 0, m0, laneHi, a0A);
    load_a_tile(xb, nidx, nmask, 0, m1, laneHi, a1A);

    for (int k = 0; k < KOFF - 1; k += 2) {
      // prefetch A for k+1, compute k
      load_a_tile(xb, nidx, nmask, k + 1, m0, laneHi, a0B);
      load_a_tile(xb, nidx, nmask, k + 1, m1, laneHi, a1B);
      conv1_step(W1t, k, laneLo, laneHi, a0A, a1A, acc0, acc1);
      // prefetch A for k+2, compute k+1
      load_a_tile(xb, nidx, nmask, k + 2, m0, laneHi, a0A);
      load_a_tile(xb, nidx, nmask, k + 2, m1, laneHi, a1A);
      conv1_step(W1t, k + 1, laneLo, laneHi, a0B, a1B, acc0, acc1);
    }
    conv1_step(W1t, KOFF - 1, laneLo, laneHi, a0A, a1A, acc0, acc1);

    // Store out1 (C/D layout: VGPR v -> row v + 8*(lane/16), col = lane%16)
    // and fold per-channel partial sums into LDS.
#pragma unroll
    for (int nt = 0; nt < 4; ++nt) {
      int col = nt * 16 + laneLo;
      float ps = 0.f, pss = 0.f;
#pragma unroll
      for (int v = 0; v < 8; ++v) {
        float f = acc0[nt][v];
        out1[(size_t)(row0 + v + laneHi * 8) * DCH + col] = f;
        ps += f; pss += f * f;
      }
      if (t1valid) {
#pragma unroll
        for (int v = 0; v < 8; ++v) {
          float f = acc1[nt][v];
          out1[(size_t)(row0 + 16 + v + laneHi * 8) * DCH + col] = f;
          ps += f; pss += f * f;
        }
      }
      atomicAdd(&lsum[col], ps);
      atomicAdd(&lssq[col], pss);
    }
  }
  __syncthreads();
  if (tid < DCH) {
    atomicAdd(&stats[tid], lsum[tid]);
    atomicAdd(&stats[DCH + tid], lssq[tid]);
  }
}

// ---------------------------------------------------------------------------
// bn_finalize: scale/shift from the global sums.
// ---------------------------------------------------------------------------
__global__ void bn_finalize(const float* __restrict__ gamma,
                            const float* __restrict__ beta,
                            float* __restrict__ stats) {
  int c = threadIdx.x;
  if (c < DCH) {
    float invN = 1.0f / (float)NPTS;
    float mean = stats[c] * invN;
    float var  = stats[DCH + c] * invN - mean * mean;
    float sc   = gamma[c] * rsqrtf(var + 1e-5f);
    stats[2 * DCH + c] = sc;
    stats[3 * DCH + c] = beta[c] - mean * sc;
  }
}

// ---------------------------------------------------------------------------
// mlp: BN -> (64->256) -> ReLU -> (256->64) + residual. Hidden activations are
// streamed in 64-wide chunks through a per-wave LDS tile (C/D layout -> A
// layout transpose), never touching HBM. 64 WMMAs per 16-row tile.
// ---------------------------------------------------------------------------
__global__ __launch_bounds__(256) void mlp_kernel(
    const float* __restrict__ out1, const float* __restrict__ x,
    const bf16_t* __restrict__ W2t, const bf16_t* __restrict__ W3t,
    const float* __restrict__ stats, float* __restrict__ out) {
  __shared__ __align__(32) bf16_t hT[WPB][16 * DCH];
  int tid = threadIdx.x;
  int wave = tid >> 5, lane = tid & 31;
  int laneLo = lane & 15, laneHi = lane >> 4;
  int tile = blockIdx.x * WPB + wave;
  if (tile >= NTILES) return;            // wave-uniform; no block barriers here
  int row0 = tile << 4;
  int m = row0 + laneLo;
  const float* scale = stats + 2 * DCH;
  const float* shift = stats + 3 * DCH;

  // Normalized A-fragments (16x64 bf16 = two K-steps of 32).
  v16bf a[2];
  const float* rp = out1 + (size_t)m * DCH;
#pragma unroll
  for (int ks = 0; ks < 2; ++ks) {
    int c0 = ks * 32 + laneHi * 16;
    v16bf av;
#pragma unroll
    for (int j = 0; j < 8; ++j)
      av[j] = (bf16_t)(rp[c0 + j] * scale[c0 + j] + shift[c0 + j]);
#pragma unroll
    for (int j = 0; j < 8; ++j)
      av[8 + j] = (bf16_t)(rp[c0 + 8 + j] * scale[c0 + 8 + j] + shift[c0 + 8 + j]);
    a[ks] = av;
  }

  v8f oacc[4];
#pragma unroll
  for (int nt = 0; nt < 4; ++nt) oacc[nt] = zero8();
  bf16_t* hrow = &hT[wave][0];

  for (int hc = 0; hc < 4; ++hc) {       // 4 hidden chunks of 64 (of 256)
    // Preload all B fragments for this chunk (conv2 and conv3) up front so
    // global loads overlap the WMMA bursts and the LDS round-trip.
    v16bf b2[8], b3[8];
#pragma unroll
    for (int j = 0; j < 4; ++j) {
      int col2 = hc * 64 + j * 16 + laneLo;
      b2[2 * j]     = *(const v16bf*)(W2t + (col2 << 6) + laneHi * 16);
      b2[2 * j + 1] = *(const v16bf*)(W2t + (col2 << 6) + 32 + laneHi * 16);
    }
#pragma unroll
    for (int ks = 0; ks < 2; ++ks) {
      int kk = hc * 2 + ks;
#pragma unroll
      for (int nt = 0; nt < 4; ++nt) {
        int col3 = nt * 16 + laneLo;
        b3[ks * 4 + nt] = *(const v16bf*)(W3t + (col3 << 8) + kk * 32 + laneHi * 16);
      }
    }
    // conv2: h[:, hc*64 .. +63] = relu(A @ W2) -> per-wave LDS tile (bf16)
#pragma unroll
    for (int j = 0; j < 4; ++j) {
      v8f h = zero8();
      h = wmma_bf16(a[0], b2[2 * j],     h);
      h = wmma_bf16(a[1], b2[2 * j + 1], h);
#pragma unroll
      for (int v = 0; v < 8; ++v)
        hrow[(v + laneHi * 8) * DCH + j * 16 + laneLo] = (bf16_t)fmaxf(h[v], 0.f);
    }
    asm volatile("s_wait_dscnt 0" ::: "memory");  // per-wave LDS store->load order

    // conv3: reload as A-fragments, accumulate over this 64-wide K chunk
#pragma unroll
    for (int ks = 0; ks < 2; ++ks) {
      int c0 = ks * 32 + laneHi * 16;
      v8bf lo = *(const v8bf*)(hrow + laneLo * DCH + c0);
      v8bf hi = *(const v8bf*)(hrow + laneLo * DCH + c0 + 8);
      v16bf a2 = concat8(lo, hi);
#pragma unroll
      for (int nt = 0; nt < 4; ++nt)
        oacc[nt] = wmma_bf16(a2, b3[ks * 4 + nt], oacc[nt]);
    }
  }

  // Residual add + store (C/D layout scatter; lanes of one VGPR cover 2x64B).
#pragma unroll
  for (int nt = 0; nt < 4; ++nt) {
    int col = nt * 16 + laneLo;
#pragma unroll
    for (int v = 0; v < 8; ++v) {
      size_t off = (size_t)(row0 + v + laneHi * 8) * DCH + col;
      out[off] = oacc[nt][v] + x[off];
    }
  }
}

// ---------------------------------------------------------------------------
extern "C" void kernel_launch(void* const* d_in, const int* in_sizes, int n_in,
                              void* d_out, int out_size, void* d_ws, size_t ws_size,
                              hipStream_t stream) {
  const float* x     = (const float*)d_in[0];
  const int*   nidx  = (const int*)d_in[1];
  const int*   nmask = (const int*)d_in[2];
  const float* W1    = (const float*)d_in[3];
  const float* gamma = (const float*)d_in[4];
  const float* beta  = (const float*)d_in[5];
  const float* W2    = (const float*)d_in[6];
  const float* W3    = (const float*)d_in[7];
  float* out = (float*)d_out;

  char* ws = (char*)d_ws;
  float*  out1 = (float*)(ws + WS_OUT1);
  bf16_t* xb   = (bf16_t*)(ws + WS_XB);
  bf16_t* W1t  = (bf16_t*)(ws + WS_W1T);
  bf16_t* W2t  = (bf16_t*)(ws + WS_W2T);
  bf16_t* W3t  = (bf16_t*)(ws + WS_W3T);
  float*  stats = (float*)(ws + WS_STAT);

  prep_x<<<(NPTS * DCH + 255) / 256, 256, 0, stream>>>(x, xb, NPTS * DCH);
  prep_w<<<(KOFF * DCH * DCH + 255) / 256, 256, 0, stream>>>(W1, W2, W3, W1t, W2t,
                                                             W3t, stats);
  conv1_kernel<<<(NPAIR + WPB - 1) / WPB, 256, 0, stream>>>(xb, nidx, nmask, W1t,
                                                            out1, stats);
  bn_finalize<<<1, 64, 0, stream>>>(gamma, beta, stats);
  mlp_kernel<<<(NTILES + WPB - 1) / WPB, 256, 0, stream>>>(out1, x, W2t, W3t,
                                                           stats, out);
}
